// Block_43903155700472
// MI455X (gfx1250) — matrix operand order repaired and checked by reading, not compiled, not run on hardware
//
#include <hip/hip_runtime.h>

// ---------------------------------------------------------------------------
// SpikeGPT / RWKV block for MI455X (gfx1250, wave32, WMMA + TDM).
// GEMMs: v_wmma_f32_16x16x32_bf16, pre-transposed bf16 weights, double-
// buffered LDS fed by TENSOR_LOAD_TO_LDS (TDM) with hardware LDS padding.
// ---------------------------------------------------------------------------

typedef __bf16 bf16_t;
typedef __bf16 v8bf  __attribute__((ext_vector_type(8)));
typedef __bf16 v16bf __attribute__((ext_vector_type(16)));
typedef float  v8f   __attribute__((ext_vector_type(8)));
typedef int    v4i   __attribute__((ext_vector_type(4)));

#define NB_   8
#define TT_   1024
#define CC_   1024
#define HH_   4096
#define NROW_ (NB_ * TT_)

#if defined(__has_builtin)
#if __has_builtin(__builtin_amdgcn_tensor_load_to_lds) && \
    __has_builtin(__builtin_amdgcn_s_wait_tensorcnt)
#define USE_TDM 1
#endif
#endif
#ifndef USE_TDM
#define USE_TDM 0
#endif

// ---------------------------- elementwise helpers --------------------------

// W[K,M] f32 -> Wt[M,K] bf16 (transpose through LDS; both sides coalesced).
__global__ __launch_bounds__(256)
void cast_transpose_k(const float* __restrict__ in, bf16_t* __restrict__ out,
                      int K, int M) {
  __shared__ bf16_t tile[32][33];
  const int lx = threadIdx.x & 31;
  const int ly = threadIdx.x >> 5;           // 0..7
#pragma unroll
  for (int i = 0; i < 32; i += 8) {
    int k = blockIdx.y * 32 + ly + i;
    int m = blockIdx.x * 32 + lx;
    tile[ly + i][lx] = (bf16_t)in[(size_t)k * M + m];
  }
  __syncthreads();
#pragma unroll
  for (int i = 0; i < 32; i += 8) {
    int m = blockIdx.x * 32 + ly + i;
    int k = blockIdx.y * 32 + lx;
    out[(size_t)m * K + k] = tile[lx][ly + i];
  }
}

__global__ __launch_bounds__(256)
void layernorm_k(const float* __restrict__ x, const float* __restrict__ sc,
                 const float* __restrict__ bi, float* __restrict__ out) {
  __shared__ float s1[256], s2[256];
  const int row = blockIdx.x;
  const float* xr = x + (size_t)row * CC_;
  float sum = 0.f, sq = 0.f;
  for (int c = threadIdx.x; c < CC_; c += 256) {
    float v = xr[c]; sum += v; sq += v * v;
  }
  s1[threadIdx.x] = sum; s2[threadIdx.x] = sq;
  __syncthreads();
  for (int o = 128; o > 0; o >>= 1) {
    if ((int)threadIdx.x < o) {
      s1[threadIdx.x] += s1[threadIdx.x + o];
      s2[threadIdx.x] += s2[threadIdx.x + o];
    }
    __syncthreads();
  }
  const float m   = s1[0] * (1.f / CC_);
  const float var = s2[0] * (1.f / CC_) - m * m;
  const float inv = rsqrtf(var + 1e-5f);
  float* orow = out + (size_t)row * CC_;
  for (int c = threadIdx.x; c < CC_; c += 256)
    orow[c] = (xr[c] - m) * inv * sc[c] + bi[c];
}

__global__ __launch_bounds__(256)
void mix3_k(const float* __restrict__ ln,
            const float* __restrict__ mk, const float* __restrict__ mv,
            const float* __restrict__ mr,
            bf16_t* __restrict__ ok, bf16_t* __restrict__ ov,
            bf16_t* __restrict__ orr) {
  size_t i = (size_t)blockIdx.x * 256 + threadIdx.x;
  int c = (int)(i % CC_);
  int t = (int)((i / CC_) % TT_);
  float cur  = ln[i];
  float prev = (t > 0) ? ln[i - CC_] : 0.f;
  float a = mk[c], b = mv[c], r = mr[c];
  ok[i]  = (bf16_t)(cur * a + prev * (1.f - a));
  ov[i]  = (bf16_t)(cur * b + prev * (1.f - b));
  orr[i] = (bf16_t)(cur * r + prev * (1.f - r));
}

__global__ __launch_bounds__(256)
void mix2_k(const float* __restrict__ ln,
            const float* __restrict__ mk, const float* __restrict__ mr,
            bf16_t* __restrict__ ok, bf16_t* __restrict__ orr) {
  size_t i = (size_t)blockIdx.x * 256 + threadIdx.x;
  int c = (int)(i % CC_);
  int t = (int)((i / CC_) % TT_);
  float cur  = ln[i];
  float prev = (t > 0) ? ln[i - CC_] : 0.f;
  float a = mk[c], r = mr[c];
  ok[i]  = (bf16_t)(cur * a + prev * (1.f - a));
  orr[i] = (bf16_t)(cur * r + prev * (1.f - r));
}

__global__ __launch_bounds__(256)
void gatemul_k(const float* __restrict__ sr, const float* __restrict__ y,
               bf16_t* __restrict__ out) {
  size_t i = (size_t)blockIdx.x * 256 + threadIdx.x;
  out[i] = (bf16_t)(sr[i] * y[i]);
}

// ---------------------------- sequential scans -----------------------------

__global__ __launch_bounds__(256)
void wkv_k(const float* __restrict__ wdec, const float* __restrict__ ufirst,
           const float* __restrict__ k, const float* __restrict__ v,
           float* __restrict__ y) {
  int idx = blockIdx.x * 256 + threadIdx.x;      // 0 .. B*C-1
  int b = idx / CC_, c = idx % CC_;
  const float W = -__expf(wdec[c]);
  const float u = ufirst[c];
  float p = 0.f, q = 0.f, o = -1e38f;
  size_t base = (size_t)b * TT_ * CC_ + c;
  for (int t = 0; t < TT_; ++t) {
    float kt = k[base], vt = v[base];
    float no = fmaxf(o, u + kt);
    float A  = __expf(o - no);
    float Bc = __expf(u + kt - no);
    y[base]  = (A * p + Bc * vt) / (A * q + Bc);
    float no2 = fmaxf(W + o, kt);
    float A2  = __expf(W + o - no2);
    float B2  = __expf(kt - no2);
    p = A2 * p + B2 * vt;
    q = A2 * q + B2;
    o = no2;
    base += CC_;
  }
}

__global__ __launch_bounds__(256)
void lif_add_k(const float* __restrict__ basex, const float* __restrict__ z,
               float* __restrict__ out) {
  int idx = blockIdx.x * 256 + threadIdx.x;      // 0 .. B*C-1
  int b = idx / CC_, c = idx % CC_;
  float vmem = 0.f;
  size_t p = (size_t)b * TT_ * CC_ + c;
  for (int t = 0; t < TT_; ++t) {
    vmem += (z[p] - vmem) * 0.5f;
    float s = (vmem >= 1.f) ? 1.f : 0.f;
    out[p] = basex[p] + s;
    vmem *= (1.f - s);
    p += CC_;
  }
}

// ---------------------------- TDM tile loader ------------------------------

#if USE_TDM
typedef unsigned tdm_u4 __attribute__((ext_vector_type(4)));
typedef int      tdm_i8 __attribute__((ext_vector_type(8)));
typedef int      tdm_i4 __attribute__((ext_vector_type(4)));

// 2D TDM copy: tile_dim0=32 (contiguous bf16), tile_dim1=128 rows,
// LDS padding: after every 16 DWORDs (one 32-elem row) insert 4 DWORDs
// -> LDS row stride = 40 bf16 (matches LDA_P/LDB_P below).
__device__ __forceinline__
void tdm_load_tile_2d(unsigned lds_byte_off, const bf16_t* gsrc,
                      int dim0, int dim1, int stride0) {
  unsigned long long ga = (unsigned long long)(const void*)gsrc;
  tdm_u4 g0;
  g0[0] = 1u;                                               // count=1 (valid)
  g0[1] = lds_byte_off;                                     // lds_addr
  g0[2] = (unsigned)ga;                                     // global_addr lo
  g0[3] = (unsigned)((ga >> 32) & 0x01FFFFFFull) | (2u << 30); // hi + type=2
  tdm_i8 g1;
  g1[0] = (1 << 16) | (1 << 20) | (3 << 22) | (3 << 25);    // 2B, pad 16dw+4dw
  g1[1] = (int)(((unsigned)dim0 & 0xFFFFu) << 16);          // dim0 lo16
  g1[2] = (int)((((unsigned)dim0 >> 16) & 0xFFFFu) |
                (((unsigned)dim1 & 0xFFFFu) << 16));        // dim0 hi | dim1 lo
  g1[3] = (int)((((unsigned)dim1 >> 16) & 0xFFFFu) |
                (32u << 16));                               // dim1 hi | tile0=32
  g1[4] = 128;                                              // tile1=128, tile2=0
  g1[5] = stride0;                                          // dim0 stride lo32
  g1[6] = 0;
  g1[7] = 0;
  tdm_i4 z4 = {0, 0, 0, 0};
  tdm_i8 z8 = {0, 0, 0, 0, 0, 0, 0, 0};
  __builtin_amdgcn_tensor_load_to_lds(g0, g1, z4, z4, z8, 0);
}
#endif

// ---------------------------- WMMA GEMM ------------------------------------
// C[N,M] = A[N,K] @ Wt[M,K]^T ; bf16 inputs, fp32 accum via
// v_wmma_f32_16x16x32_bf16. 256 threads = 8 waves, 128x128 tile, K-step 32,
// double-buffered LDS. Tiles staged by TDM (wave 0) or plain loads (fallback).
// MODE: 0 = store f32, 1 = sigmoid f32, 2 = relu^2 -> bf16, 3 = gate*acc f32.

#define GTM 128
#define GTN 128
#define GTK 32
#define LDA_P 40
#define LDB_P 40

template <int MODE, int KDIM, int MDIM>
__global__ __launch_bounds__(256)
void gemm_bf16_wmma(const bf16_t* __restrict__ A, const bf16_t* __restrict__ Wt,
                    void* __restrict__ Out, const float* __restrict__ gate) {
  __shared__ bf16_t As[2][GTM * LDA_P];
  __shared__ bf16_t Bs[2][GTN * LDB_P];

  const int tid  = threadIdx.x;
  const int wave = tid >> 5;
  const int lane = tid & 31;
  const int half = lane >> 4;
  const int ln16 = lane & 15;

  const int blockM = blockIdx.y * GTM;   // token rows
  const int blockN = blockIdx.x * GTN;   // feature cols
  const int waveRow = (wave >> 1) * 32;
  const int waveCol = (wave & 1) * 64;

  v8f acc[2][4];
#pragma unroll
  for (int r = 0; r < 2; ++r)
#pragma unroll
    for (int c = 0; c < 4; ++c)
#pragma unroll
      for (int i = 0; i < 8; ++i) acc[r][c][i] = 0.f;

#if USE_TDM
  const bf16_t* atile = A  + (size_t)blockM * KDIM;   // row blockM, k 0
  const bf16_t* btile = Wt + (size_t)blockN * KDIM;   // row blockN, k 0
  if (wave == 0) {
    tdm_load_tile_2d((unsigned)(unsigned long long)(void*)&As[0][0], atile,
                     KDIM, NROW_, KDIM);
    tdm_load_tile_2d((unsigned)(unsigned long long)(void*)&Bs[0][0], btile,
                     KDIM, MDIM, KDIM);
    __builtin_amdgcn_s_wait_tensorcnt(0);
  }
  __syncthreads();
#else
  const int arow = tid >> 1;             // 0..127
  const int aseg = (tid & 1) * 16;       // k offset 0/16
  const bf16_t* aptr = A  + (size_t)(blockM + arow) * KDIM + aseg;
  const bf16_t* bptr = Wt + (size_t)(blockN + arow) * KDIM + aseg;
  v4i ar0, ar1, br0, br1;
  ar0 = *(const v4i*)(aptr);
  ar1 = *(const v4i*)(aptr + 8);
  br0 = *(const v4i*)(bptr);
  br1 = *(const v4i*)(bptr + 8);
  *(v4i*)&As[0][arow * LDA_P + aseg]     = ar0;
  *(v4i*)&As[0][arow * LDA_P + aseg + 8] = ar1;
  *(v4i*)&Bs[0][arow * LDB_P + aseg]     = br0;
  *(v4i*)&Bs[0][arow * LDB_P + aseg + 8] = br1;
  __syncthreads();
#endif

  int buf = 0;
  for (int k0 = 0; k0 < KDIM; k0 += GTK) {
    const bool has_next = (k0 + GTK) < KDIM;

#if USE_TDM
    if (has_next && wave == 0) {   // DMA next tile into back buffer
      const int nb = buf ^ 1;
      tdm_load_tile_2d((unsigned)(unsigned long long)(void*)&As[nb][0],
                       atile + (k0 + GTK), KDIM, NROW_, KDIM);
      tdm_load_tile_2d((unsigned)(unsigned long long)(void*)&Bs[nb][0],
                       btile + (k0 + GTK), KDIM, MDIM, KDIM);
    }
#else
    if (has_next) {                // prefetch next tile into registers
      const bf16_t* an = aptr + (k0 + GTK);
      ar0 = *(const v4i*)an;
      ar1 = *(const v4i*)(an + 8);
      const bf16_t* bn = bptr + (k0 + GTK);
      br0 = *(const v4i*)bn;
      br1 = *(const v4i*)(bn + 8);
    }
#endif

    // A fragments: lane = row m, K chunks [half*8,+8) and [16+half*8,+8).
    v16bf afr[2];
#pragma unroll
    for (int r = 0; r < 2; ++r) {
      const bf16_t* ap = &As[buf][(waveRow + r * 16 + ln16) * LDA_P + half * 8];
      v8bf lo = *(const v8bf*)ap;
      v8bf hi = *(const v8bf*)(ap + 16);
      afr[r] = __builtin_shufflevector(lo, hi, 0, 1, 2, 3, 4, 5, 6, 7,
                                       8, 9, 10, 11, 12, 13, 14, 15);
    }
    // B fragments: lane = col n, K chunks [half*8,+8) and [16+half*8,+8).
    v16bf bfr[4];
#pragma unroll
    for (int c = 0; c < 4; ++c) {
      const bf16_t* bp = &Bs[buf][(waveCol + c * 16 + ln16) * LDB_P + half * 8];
      v8bf lo = *(const v8bf*)bp;
      v8bf hi = *(const v8bf*)(bp + 16);
      bfr[c] = __builtin_shufflevector(lo, hi, 0, 1, 2, 3, 4, 5, 6, 7,
                                       8, 9, 10, 11, 12, 13, 14, 15);
    }

#pragma unroll
    for (int r = 0; r < 2; ++r)
#pragma unroll
      for (int c = 0; c < 4; ++c)
        acc[r][c] = __builtin_amdgcn_wmma_f32_16x16x32_bf16(
            false, afr[r], false, bfr[c], (short)0, acc[r][c], false, false);

#if USE_TDM
    if (has_next && wave == 0) __builtin_amdgcn_s_wait_tensorcnt(0);
#else
    if (has_next) {
      const int nb = buf ^ 1;
      *(v4i*)&As[nb][arow * LDA_P + aseg]     = ar0;
      *(v4i*)&As[nb][arow * LDA_P + aseg + 8] = ar1;
      *(v4i*)&Bs[nb][arow * LDB_P + aseg]     = br0;
      *(v4i*)&Bs[nb][arow * LDB_P + aseg + 8] = br1;
    }
#endif
    __syncthreads();
    buf ^= 1;
  }

  // Epilogue: per-(r,c) base address + immediate-offset stores (MDIM const).
#pragma unroll
  for (int r = 0; r < 2; ++r) {
#pragma unroll
    for (int c = 0; c < 4; ++c) {
      const int gcol  = blockN + waveCol + c * 16 + ln16;
      const int grow0 = blockM + waveRow + r * 16 + half * 8;
      const size_t base = (size_t)grow0 * MDIM + gcol;
#pragma unroll
      for (int i = 0; i < 8; ++i) {
        float v = acc[r][c][i];
        const size_t idx = base + (size_t)i * MDIM;
        if constexpr (MODE == 0) {
          ((float*)Out)[idx] = v;
        } else if constexpr (MODE == 1) {
          ((float*)Out)[idx] = 1.f / (1.f + __expf(-v));
        } else if constexpr (MODE == 2) {
          float rl = v > 0.f ? v : 0.f;
          ((bf16_t*)Out)[idx] = (bf16_t)(rl * rl);
        } else {  // MODE == 3
          ((float*)Out)[idx] = v * gate[idx];
        }
      }
    }
  }
}

// ---------------------------- launch ---------------------------------------

extern "C" void kernel_launch(void* const* d_in, const int* in_sizes, int n_in,
                              void* d_out, int out_size, void* d_ws, size_t ws_size,
                              hipStream_t stream) {
  (void)in_sizes; (void)n_in; (void)out_size; (void)ws_size;

  const float* x      = (const float*)d_in[0];
  const float* ln1_s  = (const float*)d_in[1];
  const float* ln1_b  = (const float*)d_in[2];
  const float* ln2_s  = (const float*)d_in[3];
  const float* ln2_b  = (const float*)d_in[4];
  const float* tmk    = (const float*)d_in[5];
  const float* tmv    = (const float*)d_in[6];
  const float* tmr    = (const float*)d_in[7];
  const float* wdec   = (const float*)d_in[8];
  const float* ufirst = (const float*)d_in[9];
  const float* Wk     = (const float*)d_in[10];
  const float* Wv     = (const float*)d_in[11];
  const float* Wr     = (const float*)d_in[12];
  const float* Wo     = (const float*)d_in[13];
  const float* f_tmk  = (const float*)d_in[14];
  const float* f_tmr  = (const float*)d_in[15];
  const float* fWk    = (const float*)d_in[16];
  const float* fWr    = (const float*)d_in[17];
  const float* fWv    = (const float*)d_in[18];

  char* ws = (char*)d_ws;
  size_t off = 0;
  auto alloc = [&](size_t bytes) -> void* {
    void* p = ws + off;
    off += (bytes + 255) & ~(size_t)255;
    return p;
  };

  const size_t NC = (size_t)NROW_ * CC_;
  const size_t NH = (size_t)NROW_ * HH_;

  // Transposed bf16 weights: Wt[M][K].
  bf16_t* WkT  = (bf16_t*)alloc((size_t)CC_ * CC_ * 2);
  bf16_t* WvT  = (bf16_t*)alloc((size_t)CC_ * CC_ * 2);
  bf16_t* WrT  = (bf16_t*)alloc((size_t)CC_ * CC_ * 2);
  bf16_t* WoT  = (bf16_t*)alloc((size_t)CC_ * CC_ * 2);
  bf16_t* fWkT = (bf16_t*)alloc((size_t)CC_ * HH_ * 2);   // [H][C]
  bf16_t* fWrT = (bf16_t*)alloc((size_t)CC_ * CC_ * 2);
  bf16_t* fWvT = (bf16_t*)alloc((size_t)HH_ * CC_ * 2);   // [C][H]

  float*  lnb = (float*)alloc(NC * 4);
  bf16_t* a0  = (bf16_t*)alloc(NC * 2);
  bf16_t* a1  = (bf16_t*)alloc(NC * 2);
  bf16_t* a2  = (bf16_t*)alloc(NC * 2);
  float*  g0  = (float*)alloc(NC * 4);   // k ; reused as r2 in phase 2
  float*  g1  = (float*)alloc(NC * 4);   // v
  float*  g2  = (float*)alloc(NC * 4);   // sr
  float*  yb  = (float*)alloc(NC * 4);   // wkv output
  bf16_t* gb  = (bf16_t*)alloc(NC * 2);  // sr*y (bf16)
  float*  tm  = (float*)alloc(NC * 4);   // time-mix out ; reused as cm out
  float*  x1  = (float*)alloc(NC * 4);   // x after first residual
  bf16_t* k2  = (bf16_t*)alloc(NH * 2);  // relu(xk@fWk)^2 (bf16)
  float*  r2  = g0;
  float*  cm  = tm;

  const int ew = 256;
  const int gNC = (int)(NC / ew);        // 32768 blocks
  const int gBC = (NB_ * CC_) / ew;      // 32 blocks (scan kernels)

  // Weights -> bf16 transposed (stateless: redone every call).
  dim3 tCC(CC_ / 32, CC_ / 32);
  dim3 tCH(HH_ / 32, CC_ / 32);          // in [C,H]
  dim3 tHC(CC_ / 32, HH_ / 32);          // in [H,C]
  cast_transpose_k<<<tCC, 256, 0, stream>>>(Wk,  WkT,  CC_, CC_);
  cast_transpose_k<<<tCC, 256, 0, stream>>>(Wv,  WvT,  CC_, CC_);
  cast_transpose_k<<<tCC, 256, 0, stream>>>(Wr,  WrT,  CC_, CC_);
  cast_transpose_k<<<tCC, 256, 0, stream>>>(Wo,  WoT,  CC_, CC_);
  cast_transpose_k<<<tCH, 256, 0, stream>>>(fWk, fWkT, CC_, HH_);
  cast_transpose_k<<<tCC, 256, 0, stream>>>(fWr, fWrT, CC_, CC_);
  cast_transpose_k<<<tHC, 256, 0, stream>>>(fWv, fWvT, HH_, CC_);

  // ---- phase 1: time-mix + LIF residual ----
  layernorm_k<<<NROW_, 256, 0, stream>>>(x, ln1_s, ln1_b, lnb);
  mix3_k<<<gNC, ew, 0, stream>>>(lnb, tmk, tmv, tmr, a0, a1, a2);

  dim3 gemmC(CC_ / GTN, NROW_ / GTM);    // 8 x 64
  dim3 gemmH(HH_ / GTN, NROW_ / GTM);    // 32 x 64
  gemm_bf16_wmma<0, CC_, CC_><<<gemmC, 256, 0, stream>>>(a0, WkT, g0, nullptr);
  gemm_bf16_wmma<0, CC_, CC_><<<gemmC, 256, 0, stream>>>(a1, WvT, g1, nullptr);
  gemm_bf16_wmma<1, CC_, CC_><<<gemmC, 256, 0, stream>>>(a2, WrT, g2, nullptr);

  wkv_k<<<gBC, ew, 0, stream>>>(wdec, ufirst, g0, g1, yb);
  gatemul_k<<<gNC, ew, 0, stream>>>(g2, yb, gb);
  gemm_bf16_wmma<0, CC_, CC_><<<gemmC, 256, 0, stream>>>(gb, WoT, tm, nullptr);
  lif_add_k<<<gBC, ew, 0, stream>>>(x, tm, x1);

  // ---- phase 2: channel-mix + LIF residual ----
  layernorm_k<<<NROW_, 256, 0, stream>>>(x1, ln2_s, ln2_b, lnb);
  mix2_k<<<gNC, ew, 0, stream>>>(lnb, f_tmk, f_tmr, a0, a2);

  gemm_bf16_wmma<2, CC_, HH_><<<gemmH, 256, 0, stream>>>(a0, fWkT, k2, nullptr);
  gemm_bf16_wmma<1, CC_, CC_><<<gemmC, 256, 0, stream>>>(a2, fWrT, r2, nullptr);
  gemm_bf16_wmma<3, HH_, CC_><<<gemmC, 256, 0, stream>>>(k2, fWvT, cm, r2);

  lif_add_k<<<gBC, ew, 0, stream>>>(x1, cm, (float*)d_out);
}
